// MultiHeadAttention_71657234367050
// MI455X (gfx1250) — matrix operand order repaired and checked by reading, not compiled, not run on hardware
//
#include <hip/hip_runtime.h>

// ---------------- problem constants ----------------
constexpr int   Bc   = 4;
constexpr int   Tc   = 2048;
constexpr int   Dc   = 1024;
constexpr int   Hc   = 16;
constexpr int   HDc  = 64;
constexpr int   Mc   = Bc * Tc;          // 8192 token rows
constexpr float ATT_SCALE = 0.125f;      // 1/sqrt(64)
constexpr float LOG2E     = 1.44269504088896340736f;

typedef __bf16 bf16;
typedef __attribute__((ext_vector_type(16))) __bf16 v16bf;
typedef __attribute__((ext_vector_type(8)))  float  v8f;

#define WMMA_BF16(a, b, c) \
  __builtin_amdgcn_wmma_f32_16x16x32_bf16(false, (a), false, (b), (short)0, (c), false, false)

// ---------------- WMMA fragment loaders (contiguous k = 16h + e, same order for A and B) ----
// A (16x32 bf16, row-major src, leading dim ld): lane (m = l&15, h = l>>4),
// elems e -> k = k0 + 16h + e. One contiguous 32B chunk.
static __device__ __forceinline__ v16bf load_a_frag(const bf16* __restrict__ src,
                                                    int ld, int m0, int k0, int lane) {
  const int m  = m0 + (lane & 15);
  const int hh = lane >> 4;
  const bf16* p0 = src + (size_t)m * ld + k0 + hh * 16;
  v16bf a;
#pragma unroll
  for (int e = 0; e < 16; ++e) a[e] = p0[e];
  return a;
}

// B (32x16 bf16) given B-TRANSPOSED row-major storage srcT[n][k] (leading dim ld):
// lane (n = n0 + (l&15), h = l>>4), elems e -> k = k0 + 16h + e. One contiguous 32B chunk.
static __device__ __forceinline__ v16bf load_bt_frag(const bf16* __restrict__ srcT,
                                                     int ld, int k0, int n0, int lane) {
  const int n  = n0 + (lane & 15);
  const int hh = lane >> 4;
  const bf16* p = srcT + (size_t)n * ld + k0 + hh * 16;
  v16bf b;
#pragma unroll
  for (int e = 0; e < 16; ++e) b[e] = p[e];
  return b;
}

// 16-lane butterfly reductions; masks 1..8 stay within each half-wave,
// which is exactly one C-fragment row group on wave32.
static __device__ __forceinline__ float row_reduce_max(float v) {
  v = fmaxf(v, __shfl_xor(v, 1, 32));
  v = fmaxf(v, __shfl_xor(v, 2, 32));
  v = fmaxf(v, __shfl_xor(v, 4, 32));
  v = fmaxf(v, __shfl_xor(v, 8, 32));
  return v;
}
static __device__ __forceinline__ float row_reduce_sum(float v) {
  v += __shfl_xor(v, 1, 32);
  v += __shfl_xor(v, 2, 32);
  v += __shfl_xor(v, 4, 32);
  v += __shfl_xor(v, 8, 32);
  return v;
}

// ---------------- stage 0: fp32 -> bf16 convert ----------------
__global__ void mha_f32_to_bf16(const float* __restrict__ in, bf16* __restrict__ out, int n) {
  int i = blockIdx.x * blockDim.x + threadIdx.x;
  int stride = gridDim.x * blockDim.x;
  for (; i < n; i += stride) out[i] = (bf16)in[i];
}

// ---------------- stages 1 & 3: WMMA GEMM  C[M,N] = A[M,K] * Bt[N,K]^T -------------
// Wave tile 32(M) x 64(N), double-buffered K-slices of 32 so the 12 b128 loads of
// slice k+1 are in flight while the 8 WMMAs of slice k execute (partial s_wait_loadcnt).
// mode 0: bf16 row-major out [M][N]
// mode 1: bf16 out stored as V^T: [B][H][HD][T]
// mode 2: fp32 row-major out [M][N]
__global__ __launch_bounds__(256)
void mha_gemm_bf16(const bf16* __restrict__ A, const bf16* __restrict__ Bt,
                   void* __restrict__ Out, int M, int N, int K, int mode) {
  const int lane = threadIdx.x & 31;
  const int wave = threadIdx.x >> 5;
  const int wid  = blockIdx.x * (blockDim.x >> 5) + wave;
  const int ntiles = N / 64;
  const int mt = wid / ntiles;
  const int nt = wid - mt * ntiles;
  if (mt * 32 >= M) return;
  const int m0 = mt * 32, n0 = nt * 64;

  v8f acc[8] = {};

  // prologue: load K-slice 0
  v16bf a0 = load_a_frag(A, K, m0,      0, lane);
  v16bf a1 = load_a_frag(A, K, m0 + 16, 0, lane);
  v16bf b0 = load_bt_frag(Bt, K, 0, n0,      lane);
  v16bf b1 = load_bt_frag(Bt, K, 0, n0 + 16, lane);
  v16bf b2 = load_bt_frag(Bt, K, 0, n0 + 32, lane);
  v16bf b3 = load_bt_frag(Bt, K, 0, n0 + 48, lane);

  for (int k0 = 32; k0 < K; k0 += 32) {
    // issue next slice's loads before consuming the current one
    v16bf a0n = load_a_frag(A, K, m0,      k0, lane);
    v16bf a1n = load_a_frag(A, K, m0 + 16, k0, lane);
    v16bf b0n = load_bt_frag(Bt, K, k0, n0,      lane);
    v16bf b1n = load_bt_frag(Bt, K, k0, n0 + 16, lane);
    v16bf b2n = load_bt_frag(Bt, K, k0, n0 + 32, lane);
    v16bf b3n = load_bt_frag(Bt, K, k0, n0 + 48, lane);

    acc[0] = WMMA_BF16(a0, b0, acc[0]);
    acc[4] = WMMA_BF16(a1, b0, acc[4]);
    acc[1] = WMMA_BF16(a0, b1, acc[1]);
    acc[5] = WMMA_BF16(a1, b1, acc[5]);
    acc[2] = WMMA_BF16(a0, b2, acc[2]);
    acc[6] = WMMA_BF16(a1, b2, acc[6]);
    acc[3] = WMMA_BF16(a0, b3, acc[3]);
    acc[7] = WMMA_BF16(a1, b3, acc[7]);

    a0 = a0n; a1 = a1n; b0 = b0n; b1 = b1n; b2 = b2n; b3 = b3n;
  }
  // epilogue slice
  acc[0] = WMMA_BF16(a0, b0, acc[0]);
  acc[4] = WMMA_BF16(a1, b0, acc[4]);
  acc[1] = WMMA_BF16(a0, b1, acc[1]);
  acc[5] = WMMA_BF16(a1, b1, acc[5]);
  acc[2] = WMMA_BF16(a0, b2, acc[2]);
  acc[6] = WMMA_BF16(a1, b2, acc[6]);
  acc[3] = WMMA_BF16(a0, b3, acc[3]);
  acc[7] = WMMA_BF16(a1, b3, acc[7]);

  const int nl = lane & 15, hh = lane >> 4;
  if (mode == 0) {
    bf16* out = (bf16*)Out;
#pragma unroll
    for (int i = 0; i < 8; ++i) {
      const int col  = n0 + (i & 3) * 16 + nl;
      const int mtop = m0 + (i >> 2) * 16;
#pragma unroll
      for (int r = 0; r < 8; ++r)
        out[(size_t)(mtop + r + 8 * hh) * N + col] = (bf16)acc[i][r];
    }
  } else if (mode == 1) {      // V^T : [B][H][HD][T]
    bf16* out = (bf16*)Out;
#pragma unroll
    for (int i = 0; i < 8; ++i) {
      const int col  = n0 + (i & 3) * 16 + nl;
      const int h = col >> 6, hd = col & 63;
      const int mtop = m0 + (i >> 2) * 16;
#pragma unroll
      for (int r = 0; r < 8; ++r) {
        const int m = mtop + r + 8 * hh;
        const int b = m / Tc, t = m - b * Tc;
        out[((size_t)(b * Hc + h) * HDc + hd) * Tc + t] = (bf16)acc[i][r];
      }
    }
  } else {                     // fp32 row-major
    float* out = (float*)Out;
#pragma unroll
    for (int i = 0; i < 8; ++i) {
      const int col  = n0 + (i & 3) * 16 + nl;
      const int mtop = m0 + (i >> 2) * 16;
#pragma unroll
      for (int r = 0; r < 8; ++r)
        out[(size_t)(mtop + r + 8 * hh) * N + col] = acc[i][r];
    }
  }
}

// ---------------- stage 2: causal flash attention ----------------
// Q,K: bf16 [B][T][H*HD] row-major. Vt: bf16 [B][H][HD][T]. Y: bf16 [B][T][H*HD].
// One wave handles one (b, h, 16-query tile); 4 waves per block. K and V fragments
// for a key block are all issued up front; the softmax VALU stretch hides V latency.
__global__ __launch_bounds__(128)
void mha_attn_fwd(const bf16* __restrict__ Q, const bf16* __restrict__ Kmat,
                  const bf16* __restrict__ Vt, bf16* __restrict__ Y) {
  __shared__ bf16 ldsP[4][16 * 32];   // per-wave P staging tile (C-frag -> A-frag relayout)

  const int lane = threadIdx.x & 31;
  const int wave = threadIdx.x >> 5;
  const int wid  = blockIdx.x * 4 + wave;
  const int qt = wid & (Tc / 16 - 1);         // 128 q-tiles per (b,h)
  const int bh = wid >> 7;
  const int h  = bh & (Hc - 1);
  const int b  = bh >> 4;
  const int qbase = qt * 16;

  const bf16* qptr = Q    + (size_t)b * Tc * Dc + h * HDc;   // row t, ld = Dc
  const bf16* kptr = Kmat + (size_t)b * Tc * Dc + h * HDc;
  const bf16* vptr = Vt   + (size_t)(b * Hc + h) * HDc * Tc; // row hd, ld = Tc

  // Q tile A-fragments (HD=64 -> two K=32 chunks), kept in registers for whole loop
  const v16bf aq0 = load_a_frag(qptr, Dc, qbase, 0,  lane);
  const v16bf aq1 = load_a_frag(qptr, Dc, qbase, 32, lane);

  v8f   o[4] = {};
  float mprev[8], lsum[8];
#pragma unroll
  for (int r = 0; r < 8; ++r) { mprev[r] = -3.0e38f; lsum[r] = 0.0f; }

  const int nl = lane & 15, hh = lane >> 4;
  bf16* lp = ldsP[wave];

  const int nblk = (qbase + 16 + 31) >> 5;    // key blocks of 32, causal upper bound
  for (int kb0 = 0; kb0 < nblk; ++kb0) {
    const int kb = kb0 * 32;

    // issue ALL loads for this key block up front (K frags + V frags, one clause)
    v16bf bk0 = load_bt_frag(kptr, Dc, 0,  kb,      lane);
    v16bf bk1 = load_bt_frag(kptr, Dc, 32, kb,      lane);
    v16bf bk2 = load_bt_frag(kptr, Dc, 0,  kb + 16, lane);
    v16bf bk3 = load_bt_frag(kptr, Dc, 32, kb + 16, lane);
    v16bf bv0 = load_bt_frag(vptr, Tc, kb, 0,  lane);   // consumed after softmax:
    v16bf bv1 = load_bt_frag(vptr, Tc, kb, 16, lane);   // latency hidden by VALU work
    v16bf bv2 = load_bt_frag(vptr, Tc, kb, 32, lane);
    v16bf bv3 = load_bt_frag(vptr, Tc, kb, 48, lane);

    // S = Q * K^T  (two 16-key halves, HD contracted in two WMMAs each)
    v8f s0 = {}, s1 = {};
    s0 = WMMA_BF16(aq0, bk0, s0);
    s0 = WMMA_BF16(aq1, bk1, s0);
    s1 = WMMA_BF16(aq0, bk2, s1);
    s1 = WMMA_BF16(aq1, bk3, s1);

    // online softmax: mask, row-max, exp, rescale running state
    float corr[8];
#pragma unroll
    for (int r = 0; r < 8; ++r) {
      const int q = qbase + r + 8 * hh;       // global query index for this C-frag row
      float v0 = s0[r] * ATT_SCALE;
      float v1 = s1[r] * ATT_SCALE;
      if (kb + nl      > q) v0 = -3.0e38f;    // causal mask
      if (kb + 16 + nl > q) v1 = -3.0e38f;
      const float rm   = row_reduce_max(fmaxf(v0, v1));
      const float mnew = fmaxf(mprev[r], rm);
      const float c    = exp2f((mprev[r] - mnew) * LOG2E);
      const float p0   = exp2f((v0 - mnew) * LOG2E);
      const float p1   = exp2f((v1 - mnew) * LOG2E);
      lsum[r] = lsum[r] * c + row_reduce_sum(p0 + p1);
      mprev[r] = mnew;
      corr[r]  = c;
      s0[r] = p0; s1[r] = p1;
    }
#pragma unroll
    for (int j = 0; j < 4; ++j)
#pragma unroll
      for (int r = 0; r < 8; ++r) o[j][r] *= corr[r];

    // P (C-frag layout) -> LDS [16][32] -> A-frag layout; per-wave tile, in-order DS
#pragma unroll
    for (int r = 0; r < 8; ++r) {
      const int row = r + 8 * hh;
      lp[row * 32 + nl]      = (bf16)s0[r];
      lp[row * 32 + 16 + nl] = (bf16)s1[r];
    }
    const v16bf ap = load_a_frag(lp, 32, 0, 0, lane);

    // O += P * V
    o[0] = WMMA_BF16(ap, bv0, o[0]);
    o[1] = WMMA_BF16(ap, bv1, o[1]);
    o[2] = WMMA_BF16(ap, bv2, o[2]);
    o[3] = WMMA_BF16(ap, bv3, o[3]);
  }

  // normalize and store Y tile (bf16 row-major [B][T][D])
#pragma unroll
  for (int r = 0; r < 8; ++r) lsum[r] = 1.0f / lsum[r];
  bf16* yptr = Y + (size_t)b * Tc * Dc + h * HDc;
#pragma unroll
  for (int j = 0; j < 4; ++j)
#pragma unroll
    for (int r = 0; r < 8; ++r) {
      const int t = qbase + r + 8 * hh;
      yptr[(size_t)t * Dc + j * 16 + nl] = (bf16)(o[j][r] * lsum[r]);
    }
}

// ---------------- launch ----------------
extern "C" void kernel_launch(void* const* d_in, const int* in_sizes, int n_in,
                              void* d_out, int out_size, void* d_ws, size_t ws_size,
                              hipStream_t stream) {
  const float* x  = (const float*)d_in[0];
  const float* wq = (const float*)d_in[1];
  const float* wk = (const float*)d_in[2];
  const float* wv = (const float*)d_in[3];
  const float* wo = (const float*)d_in[4];
  float* out = (float*)d_out;

  // workspace layout (bf16 staging, 88 MiB total; fits the 192 MB L2)
  char* ws = (char*)d_ws;
  size_t off = 0;
  bf16* xb  = (bf16*)(ws + off); off += (size_t)Mc * Dc * sizeof(bf16);   // 16 MiB
  bf16* wqb = (bf16*)(ws + off); off += (size_t)Dc * Dc * sizeof(bf16);   //  2 MiB
  bf16* wkb = (bf16*)(ws + off); off += (size_t)Dc * Dc * sizeof(bf16);
  bf16* wvb = (bf16*)(ws + off); off += (size_t)Dc * Dc * sizeof(bf16);
  bf16* wob = (bf16*)(ws + off); off += (size_t)Dc * Dc * sizeof(bf16);
  bf16* qb  = (bf16*)(ws + off); off += (size_t)Mc * Dc * sizeof(bf16);   // 16 MiB
  bf16* kb  = (bf16*)(ws + off); off += (size_t)Mc * Dc * sizeof(bf16);
  bf16* vtb = (bf16*)(ws + off); off += (size_t)Mc * Dc * sizeof(bf16);   // V^T
  bf16* yb  = (bf16*)(ws + off); off += (size_t)Mc * Dc * sizeof(bf16);
  (void)in_sizes; (void)n_in; (void)out_size; (void)ws_size; (void)off;

  // stage 0: convert to bf16
  mha_f32_to_bf16<<<2048, 256, 0, stream>>>(x,  xb,  Mc * Dc);
  mha_f32_to_bf16<<<1024, 256, 0, stream>>>(wq, wqb, Dc * Dc);
  mha_f32_to_bf16<<<1024, 256, 0, stream>>>(wk, wkb, Dc * Dc);
  mha_f32_to_bf16<<<1024, 256, 0, stream>>>(wv, wvb, Dc * Dc);
  mha_f32_to_bf16<<<1024, 256, 0, stream>>>(wo, wob, Dc * Dc);

  // stage 1: Q,K,V projections. (Mc/32)*(Dc/64) = 4096 waves -> 512 blocks x 8 waves
  const int gemm_blocks = (Mc / 32) * (Dc / 64) / 8;
  mha_gemm_bf16<<<gemm_blocks, 256, 0, stream>>>(xb, wqb, qb,  Mc, Dc, Dc, 0);
  mha_gemm_bf16<<<gemm_blocks, 256, 0, stream>>>(xb, wkb, kb,  Mc, Dc, Dc, 0);
  mha_gemm_bf16<<<gemm_blocks, 256, 0, stream>>>(xb, wvb, vtb, Mc, Dc, Dc, 1);

  // stage 2: causal flash attention. B*H*(T/16) = 8192 waves -> 2048 blocks x 4 waves
  mha_attn_fwd<<<(Bc * Hc * (Tc / 16)) / 4, 128, 0, stream>>>(qb, kb, vtb, yb);

  // stage 3: output projection to fp32
  mha_gemm_bf16<<<gemm_blocks, 256, 0, stream>>>(yb, wob, out, Mc, Dc, Dc, 2);
}
